// GraphAutoencoder_39865886442299
// MI455X (gfx1250) — matrix operand order repaired and checked by reading, not compiled, not run on hardware
//
#include <hip/hip_runtime.h>
#include <hip/hip_bf16.h>

// ---------------------------------------------------------------------------
// GCN graph autoencoder on MI455X (gfx1250, wave32, WMMA).
// Dense GEMMs: v_wmma_f32_16x16x32_bf16, f32 accumulate.
//  - layer GEMMs: 1x4 output tiles per wave (A-fragment reuse x4)
//  - a_hat GEMM : 2x5 output tiles per wave (A reuse x5, B reuse x2)
// Edge aggregation: hardware f32 global atomics (agg buffers are L2-resident).
// ---------------------------------------------------------------------------

typedef __bf16 bf16_t;
typedef __attribute__((ext_vector_type(16))) __bf16 v16bf;
typedef __attribute__((ext_vector_type(8)))  __bf16 v8bf;
typedef __attribute__((ext_vector_type(8)))  float  v8f;

#define GCN_N 10000
#define GCN_E 320000
#define GCN_D 512

__device__ __forceinline__ bf16_t f2bf(float f) { return (bf16_t)f; }

__device__ __forceinline__ void atomic_add_f32(float* p, float v) {
    __hip_atomic_fetch_add(p, v, __ATOMIC_RELAXED, __HIP_MEMORY_SCOPE_AGENT);
}

// Load a 16x32 bf16 A-fragment (ISA 7.12.2 layout): lanes 0-15 hold row m,
// K {0..7,16..23}; lanes 16-31 hold K {8..15,24..31}. `p` is already offset
// to row*K + khalf*8; k0 selects the 32-wide K block.
__device__ __forceinline__ v16bf load_afrag(const bf16_t* p, int k0) {
    v8bf lo = *(const v8bf*)(p + k0);
    v8bf hi = *(const v8bf*)(p + k0 + 16);
    v16bf a;
#pragma unroll
    for (int j = 0; j < 8; ++j) { a[j] = lo[j]; a[j + 8] = hi[j]; }
    return a;
}

// Load a 32x16 bf16 B-fragment: lane holds column n = lane&15 with
// K {0..7,16..23} (lanes 16-31 take K {8..15,24..31}). `p` is already offset to
// col*K + khalf*8.
__device__ __forceinline__ v16bf load_bfrag(const bf16_t* p, int k0) {
    v8bf lo = *(const v8bf*)(p + k0);
    v8bf hi = *(const v8bf*)(p + k0 + 16);
    v16bf b;
#pragma unroll
    for (int j = 0; j < 8; ++j) { b[j] = lo[j]; b[j + 8] = hi[j]; }
    return b;
}

__device__ __forceinline__ v8f wmma_bf16(v16bf a, v16bf b, v8f c) {
    return __builtin_amdgcn_wmma_f32_16x16x32_bf16(
        false, a, false, b, (short)0, c, false, false);
}

// ---------------- degree / norm prep ----------------

__global__ void k_fill1(float* __restrict__ p, int n) {
    int i = blockIdx.x * 256 + threadIdx.x;
    if (i < n) p[i] = 1.0f;                       // self-loop baseline
}

__global__ void k_deg(const int* __restrict__ dst, float* __restrict__ deg, int e) {
    int i = blockIdx.x * 256 + threadIdx.x;
    if (i < e) atomic_add_f32(&deg[dst[i]], 1.0f);
}

__global__ void k_dis(const float* __restrict__ deg, float* __restrict__ dis, int n) {
    int i = blockIdx.x * 256 + threadIdx.x;
    if (i < n) dis[i] = rsqrtf(deg[i]);
}

__global__ void k_norm(const int* __restrict__ src, const int* __restrict__ dst,
                       const float* __restrict__ dis, float* __restrict__ nrm, int e) {
    int i = blockIdx.x * 256 + threadIdx.x;
    if (i < e) nrm[i] = dis[src[i]] * dis[dst[i]];
}

// ---------------- dtype / weight prep ----------------

__global__ void k_cvt_bf16(const float* __restrict__ in, bf16_t* __restrict__ out, int n) {
    int i = blockIdx.x * 256 + threadIdx.x;
    if (i < n) out[i] = f2bf(in[i]);
}

// W [Fin,Fout] f32 row-major -> Wt [Fout,Fin] bf16 (so WMMA B-fragments are
// two contiguous 8-K runs per lane).
__global__ void k_wt(const float* __restrict__ W, bf16_t* __restrict__ Wt, int fin, int fout) {
    int i = blockIdx.x * 256 + threadIdx.x;
    if (i >= fin * fout) return;
    int k = i / fout;
    int n = i - k * fout;
    Wt[(size_t)n * fin + k] = f2bf(W[i]);
}

// ---------------- layer GEMM: C[M,Nt] = A[M,K]*Bt[Nt,K]^T ----------------
// One 16x64 output strip (1x4 tiles) per wave; 8 waves/block. Requires
// Nt % 64 == 0 (true for Fout in {64,128,512}), M % 16 == 0, K % 32 == 0.
// All branching is wave-uniform so EXEC is all-ones around WMMA.

__global__ void k_gemm_bf16_x4(const bf16_t* __restrict__ A, const bf16_t* __restrict__ Bt,
                               float* __restrict__ C, int M, int Nt, int K) {
    const int wave   = threadIdx.x >> 5;
    const int lane   = threadIdx.x & 31;
    const int stripsN = Nt >> 6;                   // groups of 4 N-tiles
    const int strips  = (M >> 4) * stripsN;
    const int t = blockIdx.x * 8 + wave;
    if (t >= strips) return;                       // uniform per wave
    const int tm  = t / stripsN;
    const int tn0 = (t - tm * stripsN) * 4;

    const int m  = lane & 15;
    const int kh = lane >> 4;

    const bf16_t* arow = A + (size_t)(tm * 16 + m) * K + kh * 8;
    const bf16_t* brow0 = Bt + (size_t)(tn0 * 16 + m) * K + kh * 8;

    v8f acc[4] = {};
    for (int k0 = 0; k0 < K; k0 += 32) {
        v16bf a = load_afrag(arow, k0);
#pragma unroll
        for (int j = 0; j < 4; ++j) {
            v16bf b = load_bfrag(brow0 + (size_t)j * 16 * K, k0);
            acc[j] = wmma_bf16(a, b, acc[j]);
        }
    }

    // C/D layout: VGPR r -> row 16*tm + r + 8*kh, col 16*tn + (lane&15)
    float* cp = C + (size_t)(tm * 16 + kh * 8) * Nt + tn0 * 16 + m;
#pragma unroll
    for (int j = 0; j < 4; ++j)
#pragma unroll
        for (int r = 0; r < 8; ++r)
            cp[(size_t)r * Nt + j * 16] = acc[j][r];
}

// ---------------- a_hat GEMM: C = H @ H^T, K = 64 ----------------
// H is [N,64] bf16 row-major, which serves as both A (row-major) and Bt
// (column-major B == row-major H). 2x5 output tiles per wave: per K-step,
// 4 A-loads + 10 B-loads feed 10 WMMAs. 625 M-tiles is odd, so the last
// M-strip clamps its second tile (same-wave duplicate store, deterministic).

__global__ void k_ahat_gemm(const bf16_t* __restrict__ H, float* __restrict__ C,
                            int Mtiles, int Ntiles, int ldc) {
    const int wave    = threadIdx.x >> 5;
    const int lane    = threadIdx.x & 31;
    const int stripsN = Ntiles / 5;
    const int stripsM = (Mtiles + 1) >> 1;
    const int t = blockIdx.x * 8 + wave;
    if (t >= stripsM * stripsN) return;            // uniform per wave
    const int sm  = t / stripsN;
    const int sn  = t - sm * stripsN;
    const int tm0 = sm * 2;
    int tm1 = tm0 + 1; if (tm1 >= Mtiles) tm1 = Mtiles - 1;
    const int tn0 = sn * 5;

    const int m  = lane & 15;
    const int kh = lane >> 4;

    const bf16_t* a0 = H + (size_t)(tm0 * 16 + m) * 64 + kh * 8;
    const bf16_t* a1 = H + (size_t)(tm1 * 16 + m) * 64 + kh * 8;
    const bf16_t* b0 = H + (size_t)(tn0 * 16 + m) * 64 + kh * 8;

    v8f acc[2][5] = {};
#pragma unroll
    for (int k0 = 0; k0 < 64; k0 += 32) {
        v16bf af0 = load_afrag(a0, k0);
        v16bf af1 = load_afrag(a1, k0);
#pragma unroll
        for (int j = 0; j < 5; ++j) {
            v16bf b = load_bfrag(b0 + (size_t)j * 16 * 64, k0);
            acc[0][j] = wmma_bf16(af0, b, acc[0][j]);
            acc[1][j] = wmma_bf16(af1, b, acc[1][j]);
        }
    }

    float* c0 = C + (size_t)(tm0 * 16 + kh * 8) * ldc + tn0 * 16 + m;
    float* c1 = C + (size_t)(tm1 * 16 + kh * 8) * ldc + tn0 * 16 + m;
#pragma unroll
    for (int j = 0; j < 5; ++j)
#pragma unroll
        for (int r = 0; r < 8; ++r) {
            c0[(size_t)r * ldc + j * 16] = acc[0][j][r];
            c1[(size_t)r * ldc + j * 16] = acc[1][j][r];
        }
}

// ---------------- GCN aggregation ----------------

// agg[i,f] = h[i,f]*dis[i]^2 + b[f]   (self loop contribution + bias)
__global__ void k_agg_init(const float* __restrict__ h, const float* __restrict__ dis,
                           const float* __restrict__ b, float* __restrict__ agg,
                           int F, int total) {
    int i = blockIdx.x * 256 + threadIdx.x;
    if (i >= total) return;
    int row = i / F;
    int col = i - row * F;
    float d = dis[row];
    agg[i] = h[i] * d * d + b[col];
}

// per edge e, feature quad blockIdx.y: agg[dst] += h[src] * norm[e]
__global__ void k_scatter(const int* __restrict__ src, const int* __restrict__ dst,
                          const float* __restrict__ nrm, const float* __restrict__ h,
                          float* __restrict__ agg, int F, int e) {
    int i = blockIdx.x * 256 + threadIdx.x;
    if (i >= e) return;
    int s = src[i], d = dst[i];
    float w = nrm[i];
    int fo = blockIdx.y * 4;
    float4 hv = *(const float4*)(h + (size_t)s * F + fo);
    float* ap = agg + (size_t)d * F + fo;
    atomic_add_f32(ap + 0, hv.x * w);
    atomic_add_f32(ap + 1, hv.y * w);
    atomic_add_f32(ap + 2, hv.z * w);
    atomic_add_f32(ap + 3, hv.w * w);
}

// relu + dual-format writeout (bf16 for the next GEMM, f32 for x_hat)
__global__ void k_final(const float* __restrict__ agg, bf16_t* outb, float* outf, int total) {
    int i = blockIdx.x * 256 + threadIdx.x;
    if (i >= total) return;
    float v = fmaxf(agg[i], 0.0f);
    if (outb) outb[i] = f2bf(v);
    if (outf) outf[i] = v;
}

// ---------------------------------------------------------------------------

extern "C" void kernel_launch(void* const* d_in, const int* in_sizes, int n_in,
                              void* d_out, int out_size, void* d_ws, size_t ws_size,
                              hipStream_t stream) {
    (void)in_sizes; (void)n_in; (void)out_size; (void)ws_size;

    const int N = GCN_N, E = GCN_E, D = GCN_D;

    const float* x    = (const float*)d_in[0];
    const int*   edge = (const int*)d_in[1];     // [2,E]
    const int*   src  = edge;
    const int*   dst  = edge + E;
    const float* W1 = (const float*)d_in[2];  const float* b1 = (const float*)d_in[3];
    const float* W2 = (const float*)d_in[4];  const float* b2 = (const float*)d_in[5];
    const float* W3 = (const float*)d_in[6];  const float* b3 = (const float*)d_in[7];
    const float* W4 = (const float*)d_in[8];  const float* b4 = (const float*)d_in[9];
    const float* W5 = (const float*)d_in[10]; const float* b5 = (const float*)d_in[11];

    float* out = (float*)d_out;
    float* x_hat = out;                          // [N, D]
    float* a_hat = out + (size_t)N * D;          // [N, N]

    // Large f32 scratch (h and agg, each N*512) lives inside the a_hat region;
    // it is fully overwritten by the final GEMM after the layers finish.
    float* hbuf = a_hat;
    float* agg  = a_hat + (size_t)N * D;

    // Small scratch in d_ws.
    char* wp = (char*)d_ws;
    auto alloc = [&](size_t bytes) -> char* {
        char* p = wp; wp += (bytes + 255) & ~(size_t)255; return p;
    };
    float*  deg  = (float*)alloc((size_t)N * 4);
    float*  dis  = (float*)alloc((size_t)N * 4);
    float*  nrm  = (float*)alloc((size_t)E * 4);
    bf16_t* w1t  = (bf16_t*)alloc((size_t)128 * 512 * 2);
    bf16_t* w2t  = (bf16_t*)alloc((size_t)64  * 128 * 2);
    bf16_t* w3t  = (bf16_t*)alloc((size_t)128 * 64  * 2);
    bf16_t* w4t  = (bf16_t*)alloc((size_t)512 * 128 * 2);
    bf16_t* w5t  = (bf16_t*)alloc((size_t)64  * 64  * 2);
    bf16_t* xb   = (bf16_t*)alloc((size_t)N * 512 * 2);
    bf16_t* h1b  = (bf16_t*)alloc((size_t)N * 128 * 2);
    bf16_t* zb   = (bf16_t*)alloc((size_t)N * 64  * 2);
    bf16_t* h2b  = (bf16_t*)alloc((size_t)N * 128 * 2);
    bf16_t* h3b  = (bf16_t*)alloc((size_t)N * 64  * 2);

    auto cdiv = [](int a, int b) { return (a + b - 1) / b; };

    // ---- degree / norm prep ----
    k_fill1<<<cdiv(N, 256), 256, 0, stream>>>(deg, N);
    k_deg  <<<cdiv(E, 256), 256, 0, stream>>>(dst, deg, E);
    k_dis  <<<cdiv(N, 256), 256, 0, stream>>>(deg, dis, N);
    k_norm <<<cdiv(E, 256), 256, 0, stream>>>(src, dst, dis, nrm, E);
    k_cvt_bf16<<<cdiv(N * D, 256), 256, 0, stream>>>(x, xb, N * D);

    // ---- one GCN layer ----
    auto gcn = [&](const bf16_t* inb, int Fin, const float* W, const float* b,
                   bf16_t* wt, int Fout, bf16_t* outb, float* outf) {
        k_wt<<<cdiv(Fin * Fout, 256), 256, 0, stream>>>(W, wt, Fin, Fout);
        int strips = (N / 16) * (Fout / 64);
        k_gemm_bf16_x4<<<cdiv(strips, 8), 256, 0, stream>>>(inb, wt, hbuf, N, Fout, Fin);
        k_agg_init<<<cdiv(N * Fout, 256), 256, 0, stream>>>(hbuf, dis, b, agg, Fout, N * Fout);
        dim3 g(cdiv(E, 256), Fout / 4);
        k_scatter<<<g, 256, 0, stream>>>(src, dst, nrm, hbuf, agg, Fout, E);
        k_final<<<cdiv(N * Fout, 256), 256, 0, stream>>>(agg, outb, outf, N * Fout);
    };

    // encoder
    gcn(xb,  512, W1, b1, w1t, 128, h1b, nullptr);
    gcn(h1b, 128, W2, b2, w2t, 64,  zb,  nullptr);
    // attribute decoder
    gcn(zb,  64,  W3, b3, w3t, 128, h2b, nullptr);
    gcn(h2b, 128, W4, b4, w4t, 512, nullptr, x_hat);
    // structure decoder
    gcn(zb,  64,  W5, b5, w5t, 64,  h3b, nullptr);

    // a_hat = h3 @ h3^T  (Bt == row-major h3; overwrites hbuf/agg region)
    {
        int mt = N / 16, nt = N / 16;              // 625 x 625 tiles
        int strips = ((mt + 1) / 2) * (nt / 5);    // 313 * 125
        k_ahat_gemm<<<cdiv(strips, 8), 256, 0, stream>>>(h3b, a_hat, mt, nt, N);
    }
}
